// SparsePatternQuantizer_46411416600581
// MI455X (gfx1250) — compile-verified, matrix operand order checked by reading, not run on hardware
//
#include <hip/hip_runtime.h>
#include <hip/hip_bf16.h>

constexpr int A_DIM = 2048;   // atoms (K of the GEMM)
constexpr int P_DIM = 2048;   // patterns
constexpr int N_DIM = 16384;  // samples

constexpr int ROWS_PER_WG = 64;         // 4 M-tiles of 16 per wave
constexpr int LDSTRIDE = A_DIM + 8;     // ushort units; pad to avoid bank conflicts

typedef __bf16 v16bf __attribute__((ext_vector_type(16)));
typedef float  v8f   __attribute__((ext_vector_type(8)));
typedef unsigned int u32x4 __attribute__((ext_vector_type(4)));

union FragU { u32x4 u4[2]; v16bf v; };

__device__ __forceinline__ unsigned short f2bf(float f) {
  unsigned int u = __float_as_uint(f);
  u += 0x7FFFu + ((u >> 16) & 1u);   // round-to-nearest-even
  return (unsigned short)(u >> 16);
}

// ---------------- kernel: zero workspace accumulators ----------------
__global__ void k_zero(float* counts, float* sumsq) {
  int i = blockIdx.x * blockDim.x + threadIdx.x;
  if (i < P_DIM) counts[i] = 0.f;
  if (i == 0) *sumsq = 0.f;
}

// ---------------- kernel: patterns (A,P) fp32 -> patT (P,A) bf16 ----------------
__global__ __launch_bounds__(256) void k_transpose_bf16(const float* __restrict__ patterns,
                                                        unsigned short* __restrict__ patT) {
  __shared__ float tile[32][33];
  int p0 = blockIdx.x * 32, a0 = blockIdx.y * 32;
  int tx = threadIdx.x, ty = threadIdx.y;  // 32 x 8
#pragma unroll
  for (int j = 0; j < 4; ++j) {
    int a = a0 + ty + j * 8;
    tile[ty + j * 8][tx] = patterns[(size_t)a * P_DIM + p0 + tx];
  }
  __syncthreads();
#pragma unroll
  for (int j = 0; j < 4; ++j) {
    int p = p0 + ty + j * 8;
    patT[(size_t)p * A_DIM + a0 + tx] = f2bf(tile[tx][ty + j * 8]);
  }
}

// ---------------- kernel: inverse L2 norm of each pattern column ----------------
__global__ __launch_bounds__(256) void k_inv_pnorm(const float* __restrict__ patterns,
                                                   float* __restrict__ inv_p) {
  int p = blockIdx.x * blockDim.x + threadIdx.x;
  if (p >= P_DIM) return;
  float s = 0.f;
  for (int a = 0; a < A_DIM; ++a) {
    float v = patterns[(size_t)a * P_DIM + p];
    s = fmaf(v, v, s);
  }
  inv_p[p] = 1.0f / fmaxf(sqrtf(s), 1e-8f);
}

// ---------------- kernel: fused bf16-WMMA GEMM + argmax over patterns ----------------
// One workgroup (8 wave32) per 64 samples. 64 bf16 code rows staged in LDS;
// each wave covers a disjoint 256-pattern slice and computes 4 stacked M-tiles
// against each B fragment (4 WMMAs per global B load pair). B frags are
// double-buffered in registers so global loads overlap WMMA issue.
__global__ __launch_bounds__(256) void k_gemm_argmax(const float* __restrict__ codes,
                                                     const unsigned short* __restrict__ patT,
                                                     const float* __restrict__ inv_p,
                                                     float* __restrict__ out_idx_f,
                                                     int* __restrict__ out_idx_i) {
  extern __shared__ char smem[];
  unsigned short* cds = (unsigned short*)smem;  // ROWS_PER_WG * LDSTRIDE bf16
  float* redv = (float*)(smem + (size_t)ROWS_PER_WG * LDSTRIDE * sizeof(unsigned short));
  int*   redi = (int*)(redv + 8 * ROWS_PER_WG);

  const int tid = threadIdx.x;
  const int nbase = blockIdx.x * ROWS_PER_WG;

  // stage 64 code rows (transposed, bf16) into LDS: cds[nloc][a]
  {
    int nloc = tid & 63;
    int arow = tid >> 6;  // 0..3
    for (int a0 = 0; a0 < A_DIM; a0 += 4) {
      int a = a0 + arow;
      float v = codes[(size_t)a * N_DIM + nbase + nloc];
      cds[nloc * LDSTRIDE + a] = f2bf(v);
    }
  }
  __syncthreads();

  const int wave = tid >> 5;
  const int lane = tid & 31;
  const int lh  = lane >> 4;   // lane half: selects K sub-range
  const int l15 = lane & 15;   // A frag: row M ; B frag / C frag: column N

  // LDS base for each of the 4 M-tiles this wave computes
  const unsigned short* arow0 = cds + (0 * 16 + l15) * LDSTRIDE + lh * 8;
  const unsigned short* arow1 = cds + (1 * 16 + l15) * LDSTRIDE + lh * 8;
  const unsigned short* arow2 = cds + (2 * 16 + l15) * LDSTRIDE + lh * 8;
  const unsigned short* arow3 = cds + (3 * 16 + l15) * LDSTRIDE + lh * 8;

  float bestv[4][8];
  int   bestp[4][8];
#pragma unroll
  for (int m = 0; m < 4; ++m)
#pragma unroll
    for (int v = 0; v < 8; ++v) { bestv[m][v] = -3.4e38f; bestp[m][v] = 0; }

  const int pwave = wave * 256;  // this wave's pattern slice
  for (int t = 0; t < 16; ++t) {
    const int p = pwave + t * 16 + l15;
    const unsigned short* brow = patT + (size_t)p * A_DIM + lh * 8;
    __builtin_prefetch(patT + (size_t)(p + 16) * A_DIM, 0, 1);

    // double-buffered B fragments
    FragU fb[2];
    fb[0].u4[0] = *(const u32x4*)(brow);
    fb[0].u4[1] = *(const u32x4*)(brow + 16);

    v8f c0 = {}, c1 = {}, c2 = {}, c3 = {};
#pragma unroll 2
    for (int kb = 0; kb < A_DIM; kb += 32) {
      const int cur = (kb >> 5) & 1;
      const int nxt = cur ^ 1;
      // issue next step's B loads before consuming current (stays in-bounds)
      const int kpre = (kb + 32 < A_DIM) ? (kb + 32) : kb;
      fb[nxt].u4[0] = *(const u32x4*)(brow + kpre);
      fb[nxt].u4[1] = *(const u32x4*)(brow + kpre + 16);

      FragU fa0, fa1, fa2, fa3;
      fa0.u4[0] = *(const u32x4*)(arow0 + kb);
      fa0.u4[1] = *(const u32x4*)(arow0 + kb + 16);
      fa1.u4[0] = *(const u32x4*)(arow1 + kb);
      fa1.u4[1] = *(const u32x4*)(arow1 + kb + 16);
      fa2.u4[0] = *(const u32x4*)(arow2 + kb);
      fa2.u4[1] = *(const u32x4*)(arow2 + kb + 16);
      fa3.u4[0] = *(const u32x4*)(arow3 + kb);
      fa3.u4[1] = *(const u32x4*)(arow3 + kb + 16);

      c0 = __builtin_amdgcn_wmma_f32_16x16x32_bf16(false, fa0.v, false, fb[cur].v, (short)0, c0, false, false);
      c1 = __builtin_amdgcn_wmma_f32_16x16x32_bf16(false, fa1.v, false, fb[cur].v, (short)0, c1, false, false);
      c2 = __builtin_amdgcn_wmma_f32_16x16x32_bf16(false, fa2.v, false, fb[cur].v, (short)0, c2, false, false);
      c3 = __builtin_amdgcn_wmma_f32_16x16x32_bf16(false, fa3.v, false, fb[cur].v, (short)0, c3, false, false);
    }

    const float sp = inv_p[p];  // column norm (row norms can't change argmax)
#pragma unroll
    for (int v = 0; v < 8; ++v) {
      float s0 = c0[v] * sp;
      float s1 = c1[v] * sp;
      float s2 = c2[v] * sp;
      float s3 = c3[v] * sp;
      if (s0 > bestv[0][v]) { bestv[0][v] = s0; bestp[0][v] = p; }
      if (s1 > bestv[1][v]) { bestv[1][v] = s1; bestp[1][v] = p; }
      if (s2 > bestv[2][v]) { bestv[2][v] = s2; bestp[2][v] = p; }
      if (s3 > bestv[3][v]) { bestv[3][v] = s3; bestp[3][v] = p; }
    }
  }

  // reduce across the 16 lanes holding columns of the same row (per half)
#pragma unroll
  for (int off = 8; off >= 1; off >>= 1) {
#pragma unroll
    for (int m = 0; m < 4; ++m) {
#pragma unroll
      for (int v = 0; v < 8; ++v) {
        float ov = __shfl_xor(bestv[m][v], off, 16);
        int   oi = __shfl_xor(bestp[m][v], off, 16);
        if (ov > bestv[m][v] || (ov == bestv[m][v] && oi < bestp[m][v])) {
          bestv[m][v] = ov; bestp[m][v] = oi;
        }
      }
    }
  }
  if (l15 == 0) {
#pragma unroll
    for (int m = 0; m < 4; ++m) {
#pragma unroll
      for (int v = 0; v < 8; ++v) {
        int row = m * 16 + lh * 8 + v;  // C layout: M = vgpr + 8*half
        redv[wave * ROWS_PER_WG + row] = bestv[m][v];
        redi[wave * ROWS_PER_WG + row] = bestp[m][v];
      }
    }
  }
  __syncthreads();

  // combine the 8 waves' disjoint pattern slices
  if (tid < ROWS_PER_WG) {
    float bv = redv[tid]; int bi = redi[tid];
    for (int w = 1; w < 8; ++w) {
      float v = redv[w * ROWS_PER_WG + tid]; int i = redi[w * ROWS_PER_WG + tid];
      if (v > bv || (v == bv && i < bi)) { bv = v; bi = i; }
    }
    out_idx_f[nbase + tid] = (float)bi;
    out_idx_i[nbase + tid] = bi;
  }
}

// ---------------- kernel: gather quantized codes + MSE partial sums ----------------
__global__ __launch_bounds__(256) void k_quantize(const float* __restrict__ codes,
                                                  const float* __restrict__ patterns,
                                                  const int* __restrict__ idx,
                                                  float* __restrict__ out_q,
                                                  float* __restrict__ sumsq) {
  __shared__ float red[256];
  const size_t total = (size_t)A_DIM * N_DIM;
  const size_t stride = (size_t)gridDim.x * blockDim.x;
  float local = 0.f;
  for (size_t i = (size_t)blockIdx.x * blockDim.x + threadIdx.x; i < total; i += stride) {
    int n = (int)(i & (size_t)(N_DIM - 1));
    int a = (int)(i >> 14);  // N_DIM == 1<<14
    float q = patterns[(size_t)a * P_DIM + idx[n]];
    out_q[i] = q;
    float d = codes[i] - q;
    local = fmaf(d, d, local);
  }
  red[threadIdx.x] = local;
  __syncthreads();
  for (int s = 128; s > 0; s >>= 1) {
    if (threadIdx.x < s) red[threadIdx.x] += red[threadIdx.x + s];
    __syncthreads();
  }
  if (threadIdx.x == 0) atomicAdd(sumsq, red[0]);
}

// ---------------- kernel: usage histogram ----------------
__global__ void k_hist(const int* __restrict__ idx, float* __restrict__ counts) {
  int n = blockIdx.x * blockDim.x + threadIdx.x;
  if (n < N_DIM) atomicAdd(&counts[idx[n]], 1.0f);
}

// ---------------- kernel: perplexity / losses / usage ----------------
__global__ __launch_bounds__(256) void k_finalize(const float* __restrict__ counts,
                                                  const float* __restrict__ sumsq,
                                                  float* __restrict__ outs) {
  __shared__ float hred[256];
  __shared__ float ured[256];
  float h = 0.f, used = 0.f;
  for (int p = threadIdx.x; p < P_DIM; p += 256) {
    float c = counts[p];
    float pr = c * (1.0f / (float)N_DIM);
    h += pr * logf(pr + 1e-10f);
    used += (pr > 0.f) ? 1.f : 0.f;
  }
  hred[threadIdx.x] = h;
  ured[threadIdx.x] = used;
  __syncthreads();
  for (int s = 128; s > 0; s >>= 1) {
    if (threadIdx.x < s) {
      hred[threadIdx.x] += hred[threadIdx.x + s];
      ured[threadIdx.x] += ured[threadIdx.x + s];
    }
    __syncthreads();
  }
  if (threadIdx.x == 0) {
    float mse = *sumsq / ((float)A_DIM * (float)N_DIM);
    outs[0] = 0.25f * mse + mse;       // pattern_loss
    outs[1] = expf(-hred[0]);          // perplexity
    outs[2] = mse;                     // commitment_loss
    outs[3] = mse;                     // codebook_loss
    outs[4] = ured[0] / (float)P_DIM;  // pattern_usage
  }
}

extern "C" void kernel_launch(void* const* d_in, const int* in_sizes, int n_in,
                              void* d_out, int out_size, void* d_ws, size_t ws_size,
                              hipStream_t stream) {
  const float* codes    = (const float*)d_in[0];   // (A, N) fp32
  const float* patterns = (const float*)d_in[1];   // (A, P) fp32
  float* out = (float*)d_out;

  // workspace layout (~8.5 MB)
  char* ws = (char*)d_ws;
  float* inv_p  = (float*)ws;                       //  8192 B
  float* counts = (float*)(ws + 8192);              //  8192 B
  float* sumsq  = (float*)(ws + 16384);             //  4 B (padded)
  int*   idx    = (int*)(ws + 16640);               //  65536 B
  unsigned short* patT = (unsigned short*)(ws + 82176);  // P*A bf16 = 8 MB

  float* out_q       = out + N_DIM;
  float* out_scalars = out + N_DIM + (size_t)A_DIM * N_DIM;

  k_zero<<<(P_DIM + 255) / 256, 256, 0, stream>>>(counts, sumsq);
  k_transpose_bf16<<<dim3(P_DIM / 32, A_DIM / 32), dim3(32, 8), 0, stream>>>(patterns, patT);
  k_inv_pnorm<<<P_DIM / 256, 256, 0, stream>>>(patterns, inv_p);

  size_t smem = (size_t)ROWS_PER_WG * LDSTRIDE * sizeof(unsigned short)
              + (size_t)8 * ROWS_PER_WG * (sizeof(float) + sizeof(int));
  k_gemm_argmax<<<N_DIM / ROWS_PER_WG, 256, smem, stream>>>(codes, patT, inv_p, out, idx);

  k_quantize<<<8192, 256, 0, stream>>>(codes, patterns, idx, out_q, sumsq);
  k_hist<<<N_DIM / 256, 256, 0, stream>>>(idx, counts);
  k_finalize<<<1, 256, 0, stream>>>(counts, sumsq, out_scalars);
}
